// BilinearAttention_28518582845505
// MI455X (gfx1250) — matrix operand order repaired and checked by reading, not compiled, not run on hardware
//
#include <hip/hip_runtime.h>

typedef __attribute__((ext_vector_type(16))) __bf16 v16bf;
typedef __attribute__((ext_vector_type(8)))  float  v8f;
typedef unsigned int   u32;
typedef unsigned short u16;

// Dimensions (fixed by the reference)
#define Bc  8
#define SQc 2048
#define SKc 2048
#define Dc  128   // DQ == DK == DV == 128

union V16 {           // 16 bf16 operand register group for one lane
    v16bf v;
    uint4 q[2];
};

// Native f32 -> bf16 convert: gfx1250 has hardware bf16 conversion, so let the
// backend emit a single v_cvt (RNE, NaN-correct) instead of integer emulation.
__device__ __forceinline__ u16 f32_to_bf16_bits(float f) {
    union { __bf16 b; u16 u; } y;
    y.b = (__bf16)f;
    return y.u;
}

// ---------------------------------------------------------------- prep kernels
// vectorized: 4 f32 in -> 4 bf16 packed out (b64 store)
__global__ void k_cvt_bf16(const float4* __restrict__ src, uint2* __restrict__ dst, int n4) {
    int i = blockIdx.x * blockDim.x + threadIdx.x;
    int stride = gridDim.x * blockDim.x;
    for (; i < n4; i += stride) {
        float4 v = src[i];
        uint2 o;
        o.x = (u32)f32_to_bf16_bits(v.x) | ((u32)f32_to_bf16_bits(v.y) << 16);
        o.y = (u32)f32_to_bf16_bits(v.z) | ((u32)f32_to_bf16_bits(v.w) << 16);
        dst[i] = o;
    }
}

// weightsT[n][k] = weights[k][n]  (128x128, bf16)
__global__ void k_wT(const float* __restrict__ w, u16* __restrict__ wT) {
    int n = blockIdx.x, k = threadIdx.x;
    wT[n * Dc + k] = f32_to_bf16_bits(w[k * Dc + n]);
}

// vT[b][n][s] = value[b][s][n]  (bf16), coalesced 32x32 tile transpose via LDS
__global__ void k_vT(const float* __restrict__ v, u16* __restrict__ vT) {
    __shared__ u16 tile[32][33];
    int b  = blockIdx.z;
    int s0 = blockIdx.x * 32;
    int n0 = blockIdx.y * 32;
    int tx = threadIdx.x, ty = threadIdx.y;
    #pragma unroll
    for (int i = ty; i < 32; i += 8)
        tile[i][tx] = f32_to_bf16_bits(v[((size_t)b * SKc + s0 + i) * Dc + n0 + tx]);
    __syncthreads();
    #pragma unroll
    for (int i = ty; i < 32; i += 8)
        vT[((size_t)b * Dc + n0 + i) * SKc + s0 + tx] = tile[tx][i];
}

// ---------------------------------------------------------------- qw = Q @ W
// one workgroup = 16 q rows; wave w computes output columns [16w, 16w+16)
__global__ __launch_bounds__(256) void k_qw(const u16* __restrict__ qb,
                                            const u16* __restrict__ wT,
                                            u16* __restrict__ qw) {
    int rowbase = blockIdx.x * 16;
    int wave = threadIdx.x >> 5, lane = threadIdx.x & 31;
    int lh = lane & 15, hi = lane >> 4;

    V16 a[4];
    const u16* arow = qb + (size_t)(rowbase + lh) * Dc;
    #pragma unroll
    for (int c = 0; c < 4; ++c) {
        int k0 = 32 * c + 8 * hi;
        a[c].q[0] = *(const uint4*)(arow + k0);
        a[c].q[1] = *(const uint4*)(arow + k0 + 16);
    }
    int n = wave * 16 + lh;
    const u16* brow = wT + (size_t)n * Dc;
    v8f acc = {};
    #pragma unroll
    for (int c = 0; c < 4; ++c) {
        V16 bb;
        int k0 = 32 * c + 8 * hi;
        bb.q[0] = *(const uint4*)(brow + k0);
        bb.q[1] = *(const uint4*)(brow + k0 + 16);
        acc = __builtin_amdgcn_wmma_f32_16x16x32_bf16(false, a[c].v, false, bb.v,
                                                      (short)0, acc, false, false);
    }
    #pragma unroll
    for (int r = 0; r < 8; ++r)
        qw[(size_t)(rowbase + 8 * hi + r) * Dc + n] = f32_to_bf16_bits(acc[r]);
}

// ------------------------------------------- fused score + softmax + P@V
// workgroup (256 thr, 8 waves) owns (batch b, 16-row q tile).
// wave w holds scores for SK columns [256w, 256w+256) in 16 v8f accumulators.
__global__ __launch_bounds__(256) void k_attn(const u16* __restrict__ qw,
                                              const u16* __restrict__ kb,
                                              const u16* __restrict__ vT,
                                              float* __restrict__ outO,
                                              float* __restrict__ outW) {
    extern __shared__ u16 pBuf[];          // [16][SKc] bf16 probabilities (64 KB)
    __shared__ float sMax[8][16];
    __shared__ float sSum[8][16];

    int bq = blockIdx.x;
    int b  = bq >> 7;                      // SQ/16 = 128
    int qt = bq & 127;
    size_t qrow0 = (size_t)b * SQc + qt * 16;
    int wave = threadIdx.x >> 5, lane = threadIdx.x & 31;
    int lh = lane & 15, hi = lane >> 4;

    // phase-4 value^T row for this lane; prefetch it now (global_prefetch_b8)
    // so L2->WGP movement overlaps the score WMMAs + softmax below.
    const u16* vrow = vT + ((size_t)b * Dc + (wave * 16 + lh)) * SKc;
    #pragma unroll
    for (int pf = 0; pf < 4; ++pf)
        __builtin_prefetch(vrow + hi * 1024 + pf * 256, 0, 1);

    // A operand: this q tile of qw (16 x 128 bf16), kept in registers
    V16 a[4];
    const u16* arow = qw + (qrow0 + lh) * Dc;
    #pragma unroll
    for (int c = 0; c < 4; ++c) {
        int k0 = 32 * c + 8 * hi;
        a[c].q[0] = *(const uint4*)(arow + k0);
        a[c].q[1] = *(const uint4*)(arow + k0 + 16);
    }

    // ---- phase 1: scores, register resident -----------------------------
    v8f acc[16];
    v8f zero = {};
    #pragma unroll
    for (int t = 0; t < 16; ++t) acc[t] = zero;

    int col0 = wave * 256;
    #pragma unroll
    for (int t = 0; t < 16; ++t) {
        int s = col0 + t * 16 + lh;
        const u16* brow = kb + ((size_t)b * SKc + s) * Dc;
        #pragma unroll
        for (int c = 0; c < 4; ++c) {
            V16 bb;
            int k0 = 32 * c + 8 * hi;
            bb.q[0] = *(const uint4*)(brow + k0);
            bb.q[1] = *(const uint4*)(brow + k0 + 16);
            acc[t] = __builtin_amdgcn_wmma_f32_16x16x32_bf16(false, a[c].v, false, bb.v,
                                                             (short)0, acc[t], false, false);
        }
    }

    // ---- phase 2: stable softmax over SK --------------------------------
    float pm[8];
    #pragma unroll
    for (int r = 0; r < 8; ++r) {
        float m = acc[0][r];
        #pragma unroll
        for (int t = 1; t < 16; ++t) m = fmaxf(m, acc[t][r]);
        #pragma unroll
        for (int mask = 1; mask < 16; mask <<= 1) m = fmaxf(m, __shfl_xor(m, mask, 32));
        pm[r] = m;
    }
    if (lh == 0) {
        #pragma unroll
        for (int r = 0; r < 8; ++r) sMax[wave][8 * hi + r] = pm[r];
    }
    __syncthreads();
    float rowmax[8];
    #pragma unroll
    for (int r = 0; r < 8; ++r) {
        float m = sMax[0][8 * hi + r];
        #pragma unroll
        for (int w = 1; w < 8; ++w) m = fmaxf(m, sMax[w][8 * hi + r]);
        rowmax[r] = m;
    }

    float ps[8];
    #pragma unroll
    for (int r = 0; r < 8; ++r) ps[r] = 0.f;
    #pragma unroll
    for (int t = 0; t < 16; ++t) {
        #pragma unroll
        for (int r = 0; r < 8; ++r) {
            float e = __expf(acc[t][r] - rowmax[r]);
            acc[t][r] = e;
            ps[r] += e;
        }
    }
    #pragma unroll
    for (int r = 0; r < 8; ++r) {
        #pragma unroll
        for (int mask = 1; mask < 16; mask <<= 1) ps[r] += __shfl_xor(ps[r], mask, 32);
    }
    if (lh == 0) {
        #pragma unroll
        for (int r = 0; r < 8; ++r) sSum[wave][8 * hi + r] = ps[r];
    }
    __syncthreads();
    float inv[8];
    #pragma unroll
    for (int r = 0; r < 8; ++r) {
        float s = sSum[0][8 * hi + r];
        #pragma unroll
        for (int w = 1; w < 8; ++w) s += sSum[w][8 * hi + r];
        inv[r] = 1.f / s;
    }

    // ---- phase 3: single HBM write of weight; bf16 P into LDS -----------
    #pragma unroll
    for (int t = 0; t < 16; ++t) {
        int n = col0 + t * 16 + lh;
        #pragma unroll
        for (int r = 0; r < 8; ++r) {
            int m = 8 * hi + r;
            float wv = acc[t][r] * inv[r];
            outW[(qrow0 + m) * (size_t)SKc + n] = wv;
            pBuf[m * SKc + n] = f32_to_bf16_bits(wv);
        }
    }
    __syncthreads();

    // ---- phase 4: O = P @ V (wave w -> DV columns [16w,16w+16)) ---------
    v8f o = zero;
    int n = wave * 16 + lh;
    const u16* prow = pBuf + lh * SKc;
    #pragma unroll 4
    for (int kk = 0; kk < 64; ++kk) {
        V16 ap, bv;
        int ka = 32 * kk + 8 * hi;
        ap.q[0] = *(const uint4*)(prow + ka);
        ap.q[1] = *(const uint4*)(prow + ka + 16);
        int kb0 = 32 * kk + 8 * hi;
        bv.q[0] = *(const uint4*)(vrow + kb0);
        bv.q[1] = *(const uint4*)(vrow + kb0 + 16);
        o = __builtin_amdgcn_wmma_f32_16x16x32_bf16(false, ap.v, false, bv.v,
                                                    (short)0, o, false, false);
    }
    #pragma unroll
    for (int r = 0; r < 8; ++r)
        outO[(qrow0 + 8 * hi + r) * (size_t)Dc + n] = o[r];
}

// ---------------------------------------------------------------- launcher
extern "C" void kernel_launch(void* const* d_in, const int* in_sizes, int n_in,
                              void* d_out, int out_size, void* d_ws, size_t ws_size,
                              hipStream_t stream) {
    (void)in_sizes; (void)n_in; (void)out_size; (void)ws_size;
    const float* query   = (const float*)d_in[0];
    const float* key     = (const float*)d_in[1];
    const float* value   = (const float*)d_in[2];
    const float* weights = (const float*)d_in[3];

    char* ws = (char*)d_ws;
    size_t off = 0;
    u16* qb = (u16*)(ws + off); off += (size_t)Bc * SQc * Dc * 2;   // query  bf16
    u16* kb = (u16*)(ws + off); off += (size_t)Bc * SKc * Dc * 2;   // key    bf16
    u16* vT = (u16*)(ws + off); off += (size_t)Bc * Dc * SKc * 2;   // value^T bf16
    u16* wT = (u16*)(ws + off); off += (size_t)Dc * Dc * 2;         // weights^T bf16
    u16* qw = (u16*)(ws + off);                                     // Q@W    bf16

    float* outO = (float*)d_out;                       // [B,SQ,DV]
    float* outW = outO + (size_t)Bc * SQc * Dc;        // [B,SQ,SK]

    k_cvt_bf16<<<1024, 256, 0, stream>>>((const float4*)query, (uint2*)qb, (Bc * SQc * Dc) / 4);
    k_cvt_bf16<<<1024, 256, 0, stream>>>((const float4*)key,   (uint2*)kb, (Bc * SKc * Dc) / 4);
    dim3 gt(SKc / 32, Dc / 32, Bc), bt(32, 8);
    k_vT<<<gt, bt, 0, stream>>>(value, vT);
    k_wT<<<Dc, Dc, 0, stream>>>(weights, wT);

    k_qw  <<<(Bc * SQc) / 16, 256, 0, stream>>>(qb, wT, qw);
    k_attn<<<(Bc * SQc) / 16, 256, 16 * SKc * sizeof(u16), stream>>>(qw, kb, vT, outO, outW);
}
